// DQN_GNN_66460323938695
// MI455X (gfx1250) — matrix operand order repaired and checked by reading, not compiled, not run on hardware
//
#include <hip/hip_runtime.h>
#include <hip/hip_bf16.h>

// ---------------------------------------------------------------------------
// CDNA5 WMMA types (wave32): v_wmma_f32_16x16x32_bf16
// ---------------------------------------------------------------------------
typedef __attribute__((ext_vector_type(16))) __bf16 v16bf;
typedef __attribute__((ext_vector_type(8)))  float  v8f;

#define N_NODES_C 20000
#define N_EDGES_C 160000
#define N_GRAPHS_C 64
#define IN_DIM_C 128
#define HID_C 512
#define N_ACT_C 32

__device__ __forceinline__ unsigned short f2bf(float f) {
    // round-to-nearest-even fp32 -> bf16
    unsigned int u = __float_as_uint(f);
    unsigned int r = u + 0x7FFFu + ((u >> 16) & 1u);
    return (unsigned short)(r >> 16);
}

// ---------------------------------------------------------------------------
// WMMA GEMM on pre-converted bf16 operands.
//   A  : bf16 row-major [M,K]
//   Bt : bf16 TRANSPOSED [N,K]   (Bt[n*K+k] = B[k*N+n])
//   C  : fp32 [M,N] or bf16 [M,N] (OUT_BF16)
// Wave tile: 16 rows x (16*NT) cols; A fragment reused across NT WMMAs.
// Block = 256 threads = 8 waves (fills one WGP 2x). K%32==0, N%(16*NT)==0.
// All NT B fragments are loaded into distinct registers BEFORE the WMMAs so
// the scheduler can overlap v_wmma with outstanding global_load_b128s.
// ---------------------------------------------------------------------------
union FragAB { uint4 q[2]; v16bf v; };

template<int NT, int OUT_BF16>
__global__ __launch_bounds__(256)
void wmma_gemm_bf16(const unsigned short* __restrict__ A,
                    const unsigned short* __restrict__ Bt,
                    const float* __restrict__ bias, void* __restrict__ Cv,
                    int M, int N, int K, int relu) {
    const int wave   = threadIdx.x >> 5;
    const int lane   = threadIdx.x & 31;
    const int mtile  = blockIdx.x * 8 + wave;
    if (mtile * 16 >= M) return;                  // wave-uniform; EXEC all-1
    const int ntile0 = blockIdx.y * NT;
    const int l16    = lane & 15;
    const int khalf  = lane >> 4;

    const int m = mtile * 16 + l16;
    const unsigned short* __restrict__ Arow = A + (size_t)m * K;

    // Per-lane base pointers for the NT B-columns (32 contiguous bf16 / step)
    const unsigned short* Brow[NT];
    #pragma unroll
    for (int t = 0; t < NT; ++t) {
        int n_t = (ntile0 + t) * 16 + l16;
        Brow[t] = Bt + (size_t)n_t * K + khalf * 16;
    }

    v8f acc[NT];
    #pragma unroll
    for (int t = 0; t < NT; ++t) acc[t] = (v8f){};

    for (int k0 = 0; k0 < K; k0 += 32) {
        FragAB a, b[NT];
        // Issue ALL loads first (distinct registers -> staggered waits)
        a.q[0] = *(const uint4*)(Arow + k0 + khalf * 16);
        a.q[1] = *(const uint4*)(Arow + k0 + khalf * 16 + 8);
        #pragma unroll
        for (int t = 0; t < NT; ++t) {
            b[t].q[0] = *(const uint4*)(Brow[t] + k0);
            b[t].q[1] = *(const uint4*)(Brow[t] + k0 + 8);
        }
        // Then the NT matrix ops, reusing the A fragment
        #pragma unroll
        for (int t = 0; t < NT; ++t) {
            acc[t] = __builtin_amdgcn_wmma_f32_16x16x32_bf16(
                false, a.v, false, b[t].v, (short)0, acc[t], false, false);
        }
    }

    // C/D layout: element r -> row (r + 8*khalf), col l16 of the tile
    const int rowbase = mtile * 16 + (khalf ? 8 : 0);
    #pragma unroll
    for (int t = 0; t < NT; ++t) {
        int n_t = (ntile0 + t) * 16 + l16;
        float bn = bias ? bias[n_t] : 0.0f;
        #pragma unroll
        for (int r = 0; r < 8; ++r) {
            float v = acc[t][r] + bn;
            if (relu) v = fmaxf(v, 0.0f);
            if (OUT_BF16)
                ((unsigned short*)Cv)[(size_t)(rowbase + r) * N + n_t] = f2bf(v);
            else
                ((float*)Cv)[(size_t)(rowbase + r) * N + n_t] = v;
        }
    }
}

// ---------------------------------------------------------------------------
// One-time converts (per call; deterministic)
// ---------------------------------------------------------------------------
__global__ void f32_to_bf16(const float* __restrict__ s, unsigned short* __restrict__ d, int n) {
    int t = blockIdx.x * blockDim.x + threadIdx.x;
    if (t < n) d[t] = f2bf(s[t]);
}
// Wt[n*K+k] = bf16(W[k*N+n])
__global__ void wtrans_bf16(const float* __restrict__ W, unsigned short* __restrict__ Wt,
                            int K, int N) {
    int t = blockIdx.x * blockDim.x + threadIdx.x;
    if (t >= K * N) return;
    int n = t / K, k = t - n * K;
    Wt[t] = f2bf(W[(size_t)k * N + n]);
}

// ---------------------------------------------------------------------------
// Degree / normalization
// ---------------------------------------------------------------------------
__global__ void deg_init(float* __restrict__ d, int n) {
    int t = blockIdx.x * blockDim.x + threadIdx.x;
    if (t < n) d[t] = 1.0f;                      // self loop
}
__global__ void deg_edges(float* __restrict__ d, const int* __restrict__ dst, int E) {
    int t = blockIdx.x * blockDim.x + threadIdx.x;
    if (t < E) atomicAdd(&d[dst[t]], 1.0f);
}
__global__ void deg_rsqrt(float* __restrict__ d, int n) {
    int t = blockIdx.x * blockDim.x + threadIdx.x;
    if (t < n) d[t] = rsqrtf(d[t]);
}

__global__ void fill_zero4(float4* __restrict__ p, int n4) {
    int t = blockIdx.x * blockDim.x + threadIdx.x;
    if (t < n4) p[t] = make_float4(0.f, 0.f, 0.f, 0.f);
}

// ---------------------------------------------------------------------------
// Edge scatter: agg[dst] += h[src] * dinv[src]*dinv[dst]   (fp32, L2 atomics)
// ---------------------------------------------------------------------------
__global__ void edge_scatter(const float* __restrict__ h, float* __restrict__ agg,
                             const int* __restrict__ src, const int* __restrict__ dst,
                             const float* __restrict__ dinv, int E, int F) {
    int t = blockIdx.x * blockDim.x + threadIdx.x;
    const int chunks = F >> 2;
    int e = t / chunks;
    if (e >= E) return;
    int c = (t - e * chunks) << 2;
    int s = src[e], d = dst[e];
    float norm = dinv[s] * dinv[d];
    const float4 hv = *(const float4*)(h + (size_t)s * F + c);
    float* out = agg + (size_t)d * F + c;
    atomicAdd(out + 0, hv.x * norm);
    atomicAdd(out + 1, hv.y * norm);
    atomicAdd(out + 2, hv.z * norm);
    atomicAdd(out + 3, hv.w * norm);
}

// ---------------------------------------------------------------------------
// Layer epilogue: xbf = bf16( relu(agg + h*dinv^2 + bias) )
// ---------------------------------------------------------------------------
__global__ void gcn_finalize(const float* __restrict__ agg, const float* __restrict__ h,
                             const float* __restrict__ dinv, const float* __restrict__ bias,
                             unsigned short* __restrict__ xbf, int Nn, int F) {
    int t = blockIdx.x * blockDim.x + threadIdx.x;
    const int chunks = F >> 2;
    if (t >= Nn * chunks) return;
    int i = t / chunks;
    int c = (t - i * chunks) << 2;
    float di = dinv[i];
    float w = di * di;
    float4 a  = *(const float4*)(agg + (size_t)i * F + c);
    float4 hv = *(const float4*)(h + (size_t)i * F + c);
    float x0 = fmaxf(a.x + hv.x * w + bias[c + 0], 0.0f);
    float x1 = fmaxf(a.y + hv.y * w + bias[c + 1], 0.0f);
    float x2 = fmaxf(a.z + hv.z * w + bias[c + 2], 0.0f);
    float x3 = fmaxf(a.w + hv.w * w + bias[c + 3], 0.0f);
    uint2 pk;
    pk.x = (unsigned)f2bf(x0) | ((unsigned)f2bf(x1) << 16);
    pk.y = (unsigned)f2bf(x2) | ((unsigned)f2bf(x3) << 16);
    *(uint2*)(xbf + (size_t)i * F + c) = pk;
}

// ---------------------------------------------------------------------------
// Global max pool over bf16 activations (>=0 post-ReLU -> bit compare exact)
// ---------------------------------------------------------------------------
__global__ void pool_max(const unsigned short* __restrict__ xbf, const int* __restrict__ batch,
                         unsigned int* __restrict__ pooled, int Nn, int F) {
    int t = blockIdx.x * blockDim.x + threadIdx.x;
    if (t >= Nn * F) return;
    int i = t / F;
    int f = t - i * F;
    int g = batch[i];
    unsigned int v = ((unsigned int)xbf[t]) << 16;   // bf16 -> fp32 bits
    atomicMax(&pooled[(size_t)g * F + f], v);
}

// ---------------------------------------------------------------------------
// Launch
// ---------------------------------------------------------------------------
extern "C" void kernel_launch(void* const* d_in, const int* in_sizes, int n_in,
                              void* d_out, int out_size, void* d_ws, size_t ws_size,
                              hipStream_t stream) {
    (void)in_sizes; (void)n_in; (void)out_size; (void)ws_size;

    const float* tree_x = (const float*)d_in[0];
    const int*   eidx   = (const int*)d_in[1];
    const int*   batch  = (const int*)d_in[2];
    const float* W1 = (const float*)d_in[3];  const float* b1 = (const float*)d_in[4];
    const float* W2 = (const float*)d_in[5];  const float* b2 = (const float*)d_in[6];
    const float* W3 = (const float*)d_in[7];  const float* b3 = (const float*)d_in[8];
    const float* Wf1 = (const float*)d_in[9];  const float* bf1 = (const float*)d_in[10];
    const float* Wf2 = (const float*)d_in[11]; const float* bf2 = (const float*)d_in[12];
    const float* Wf3 = (const float*)d_in[13]; const float* bf3 = (const float*)d_in[14];
    const float* Wo  = (const float*)d_in[15]; const float* bo  = (const float*)d_in[16];

    const int Nn = N_NODES_C, E = N_EDGES_C, G = N_GRAPHS_C, F = HID_C;
    const int* src = eidx;
    const int* dst = eidx + E;

    // -------- workspace carve (256B-aligned regions) --------
    const size_t NF = (size_t)Nn * F;                 // 10,240,000
    char* p = (char*)d_ws;
    auto carve = [&](size_t bytes) {
        char* r = p;
        p += (bytes + 255) & ~(size_t)255;
        return r;
    };
    float* h    = (float*)carve(NF * 4);              // fp32 x@W
    float* agg  = (float*)carve(NF * 4);              // fp32 scatter target
    unsigned short* xbf   = (unsigned short*)carve(NF * 2);            // bf16 activations
    unsigned short* xinbf = (unsigned short*)carve((size_t)Nn * IN_DIM_C * 2);
    float* dinv   = (float*)carve((size_t)Nn * 4);
    float* pooled = (float*)carve((size_t)G * F * 4);
    unsigned short* pbf = (unsigned short*)carve((size_t)G * F * 2);
    unsigned short* mA  = (unsigned short*)carve((size_t)G * F * 2);
    unsigned short* mB  = (unsigned short*)carve((size_t)G * F * 2);
    unsigned short* W1t = (unsigned short*)carve((size_t)IN_DIM_C * F * 2);
    unsigned short* W2t = (unsigned short*)carve((size_t)F * F * 2);
    unsigned short* W3t = (unsigned short*)carve((size_t)F * F * 2);
    unsigned short* Wf1t = (unsigned short*)carve((size_t)F * F * 2);
    unsigned short* Wf2t = (unsigned short*)carve((size_t)F * F * 2);
    unsigned short* Wf3t = (unsigned short*)carve((size_t)F * F * 2);
    unsigned short* Wot  = (unsigned short*)carve((size_t)F * N_ACT_C * 2);

    const int TB = 256;
    auto blocks = [](long long n, int tb) { return (unsigned)((n + tb - 1) / tb); };

    // -------- weight convert+transpose (bf16, N x K) --------
    wtrans_bf16<<<blocks((long long)IN_DIM_C * F, TB), TB, 0, stream>>>(W1, W1t, IN_DIM_C, F);
    wtrans_bf16<<<blocks((long long)F * F, TB), TB, 0, stream>>>(W2, W2t, F, F);
    wtrans_bf16<<<blocks((long long)F * F, TB), TB, 0, stream>>>(W3, W3t, F, F);
    wtrans_bf16<<<blocks((long long)F * F, TB), TB, 0, stream>>>(Wf1, Wf1t, F, F);
    wtrans_bf16<<<blocks((long long)F * F, TB), TB, 0, stream>>>(Wf2, Wf2t, F, F);
    wtrans_bf16<<<blocks((long long)F * F, TB), TB, 0, stream>>>(Wf3, Wf3t, F, F);
    wtrans_bf16<<<blocks((long long)F * N_ACT_C, TB), TB, 0, stream>>>(Wo, Wot, F, N_ACT_C);

    // -------- input features -> bf16 --------
    f32_to_bf16<<<blocks((long long)Nn * IN_DIM_C, TB), TB, 0, stream>>>(tree_x, xinbf, Nn * IN_DIM_C);

    // -------- degrees -> dinv --------
    deg_init <<<blocks(Nn, TB), TB, 0, stream>>>(dinv, Nn);
    deg_edges<<<blocks(E, TB),  TB, 0, stream>>>(dinv, dst, E);
    deg_rsqrt<<<blocks(Nn, TB), TB, 0, stream>>>(dinv, Nn);

    auto gemm_f32out = [&](const unsigned short* A, const unsigned short* Bt,
                           int M, int Ncols, int K) {
        dim3 grid(((M + 15) / 16 + 7) / 8, Ncols / 64);
        wmma_gemm_bf16<4, 0><<<grid, 256, 0, stream>>>(A, Bt, nullptr, h, M, Ncols, K, 0);
    };
    auto gcn_layer = [&](const unsigned short* xin, const unsigned short* Wt,
                         const float* bias, int K) {
        gemm_f32out(xin, Wt, Nn, F, K);                               // h = x @ W
        fill_zero4<<<blocks((long long)(NF >> 2), TB), TB, 0, stream>>>((float4*)agg, (int)(NF >> 2));
        edge_scatter<<<blocks((long long)E * (F >> 2), TB), TB, 0, stream>>>(h, agg, src, dst, dinv, E, F);
        gcn_finalize<<<blocks((long long)Nn * (F >> 2), TB), TB, 0, stream>>>(agg, h, dinv, bias, xbf, Nn, F);
    };

    // -------- 3 GCN layers (xbf reused: consumed by gemm before overwrite) --------
    gcn_layer(xinbf, W1t, b1, IN_DIM_C);
    gcn_layer(xbf,   W2t, b2, HID_C);
    gcn_layer(xbf,   W3t, b3, HID_C);

    // -------- global max pool --------
    fill_zero4<<<blocks((long long)(G * F) >> 2, TB), TB, 0, stream>>>((float4*)pooled, (G * F) >> 2);
    pool_max<<<blocks((long long)Nn * F, TB), TB, 0, stream>>>(xbf, batch, (unsigned int*)pooled, Nn, F);
    f32_to_bf16<<<blocks((long long)G * F, TB), TB, 0, stream>>>(pooled, pbf, G * F);

    // -------- MLP head --------
    {
        dim3 grid(1, F / 64);   // M=64 -> 4 M-tiles = 4 waves in one block
        wmma_gemm_bf16<4, 1><<<grid, 256, 0, stream>>>(pbf, Wf1t, bf1, mA, G, F, F, 1);
        wmma_gemm_bf16<4, 1><<<grid, 256, 0, stream>>>(mA,  Wf2t, bf2, mB, G, F, F, 1);
        wmma_gemm_bf16<4, 1><<<grid, 256, 0, stream>>>(mB,  Wf3t, bf3, mA, G, F, F, 1);
    }
    {
        dim3 grid(1, N_ACT_C / 32);  // NT=2 -> 32 cols per wave-tile group
        wmma_gemm_bf16<2, 0><<<grid, 256, 0, stream>>>(mA, Wot, bo, (float*)d_out, G, N_ACT_C, HID_C, 0);
    }
}
